// DCGRUCell_5574867550587
// MI455X (gfx1250) — matrix operand order repaired and checked by reading, not compiled, not run on hardware
//
#include <hip/hip_runtime.h>
#include <math.h>

// ---------------- problem constants ----------------
#define NN      8000          // nodes
#define DEG     8
#define BSZ     32            // batch
#define IDIM    2
#define UNITS   64
#define FDIM    (IDIM + UNITS)        // 66
#define MMAT    5                      // 2*K+1
#define FB      (FDIM * BSZ)           // 2112 columns of diffusion state
#define FM      (FDIM * MMAT)          // 330  (GEMM K)
#define KPAD    352                    // 11 * 32
#define KTILES  (KPAD / 32)            // 11
#define NNZ     (NN * DEG)             // 64000
#define ROWS    (BSZ * NN)             // 256000 GEMM rows

typedef __attribute__((ext_vector_type(16))) __bf16 v16bf;
typedef __attribute__((ext_vector_type(8)))  float  v8f;

// ================= CSR build for support0 =================
__global__ void count_kernel(const int* __restrict__ rows, int* __restrict__ cnt) {
    int i = blockIdx.x * 256 + threadIdx.x;
    if (i < NNZ) atomicAdd(&cnt[rows[i]], 1);
}

__global__ __launch_bounds__(1024) void scan_kernel(const int* __restrict__ cnt,
                                                    int* __restrict__ rptr) {
    __shared__ int part[1024];
    int t = threadIdx.x;
    int local[8];
    int s = 0;
    #pragma unroll
    for (int q = 0; q < 8; ++q) {
        int i = t * 8 + q;
        int v = (i < NN) ? cnt[i] : 0;
        local[q] = s;            // exclusive within thread
        s += v;
    }
    part[t] = s;
    __syncthreads();
    for (int off = 1; off < 1024; off <<= 1) {
        int v = (t >= off) ? part[t - off] : 0;
        __syncthreads();
        part[t] += v;
        __syncthreads();
    }
    int base = (t > 0) ? part[t - 1] : 0;   // exclusive block prefix
    #pragma unroll
    for (int q = 0; q < 8; ++q) {
        int i = t * 8 + q;
        if (i < NN) rptr[i] = base + local[q];
    }
    if (t == 1023) rptr[NN] = part[1023];
}

__global__ void fill_kernel(const int* __restrict__ rows, const int* __restrict__ cols,
                            const float* __restrict__ vals, const int* __restrict__ rptr,
                            int* __restrict__ cur, int* __restrict__ ccols,
                            float* __restrict__ cvals) {
    int i = blockIdx.x * 256 + threadIdx.x;
    if (i >= NNZ) return;
    int r = rows[i];
    int p = rptr[r] + atomicAdd(&cur[r], 1);
    ccols[p] = cols[i];
    cvals[p] = vals[i];
}

// ====== weight permute + pad + pre-swizzle into WMMA B-fragment layout =========
// B-fragment (16x16x32 bf16): lane L holds col n = L&15, k = 8*(L>=16) + (e&7) + 16*(e>=8).
// Stored per (k-tile kt, n-tile nt): WbF[(((kt*NT)+nt)*32 + L)*16 + e]
// -> each lane reads 32B contiguous (two b128). k is permuted k' = m*66 + f;
// original weight row = f*5 + m. Zero pad for k' >= 330.
template <int OC>
__global__ void convW_kernel(const float* __restrict__ W, __bf16* __restrict__ WbF) {
    constexpr int NT = OC / 16;
    constexpr int TOTAL = KTILES * NT * 32 * 16;
    int idx = blockIdx.x * 256 + threadIdx.x;
    if (idx >= TOTAL) return;
    int e  = idx & 15;
    int L  = (idx >> 4) & 31;
    int nt = (idx >> 9) % NT;
    int kt = idx / (512 * NT);
    int k  = kt * 32 + ((L >> 4) << 3) + (e & 7) + ((e >> 3) << 4);
    int n  = nt * 16 + (L & 15);
    float v = 0.f;
    if (k < FM) {
        int m = k / FDIM, f = k - m * FDIM;
        v = W[(f * MMAT + m) * OC + n];
    }
    WbF[idx] = (__bf16)v;
}

// ================= diffusion-state builders =================
// layout: x0[n*FB + b*66 + f]  (b-major so GEMM k-loads are unit-stride)
__global__ void build_x0_a(const float* __restrict__ inputs, const float* __restrict__ hx,
                           float* __restrict__ x0) {
    int idx = blockIdx.x * 256 + threadIdx.x;     // NN*FB = 16,896,000 exactly
    int n = idx / FB, c = idx - n * FB;
    int b = c / FDIM, f = c - b * FDIM;
    float v = (f < IDIM) ? inputs[b * (NN * IDIM) + n * IDIM + f]
                         : hx[b * (NN * UNITS) + n * UNITS + (f - IDIM)];
    x0[idx] = v;
}

// state = r * hx, with r = sigmoid(value[..., :UNITS]) already applied in GEMM1 epilogue
__global__ void build_x0_b(const float* __restrict__ inputs, const float* __restrict__ value,
                           const float* __restrict__ hx, float* __restrict__ x0) {
    int idx = blockIdx.x * 256 + threadIdx.x;
    int n = idx / FB, c = idx - n * FB;
    int b = c / FDIM, f = c - b * FDIM;
    float v;
    if (f < IDIM) {
        v = inputs[b * (NN * IDIM) + n * IDIM + f];
    } else {
        int j = f - IDIM;
        float r = value[(b * NN + n) * (2 * UNITS) + j];
        v = r * hx[b * (NN * UNITS) + n * UNITS + j];
    }
    x0[idx] = v;
}

// ================= SPMM (gather form): y = alpha*(S@x) [- zsub] =================
// block = 352 threads (11 waves), one node-row per block, 6 strided cols per thread
__global__ __launch_bounds__(352) void spmm_kernel(const int* __restrict__ rptr,
                                                   const int* __restrict__ cidx,
                                                   const float* __restrict__ cvals,
                                                   const float* __restrict__ x,
                                                   const float* __restrict__ zsub,
                                                   float alpha, float* __restrict__ y) {
    int r = blockIdx.x;
    int t = threadIdx.x;
    float acc[6] = {0.f, 0.f, 0.f, 0.f, 0.f, 0.f};
    int s, e;
    if (rptr) { s = rptr[r]; e = rptr[r + 1]; }
    else      { s = r * DEG; e = s + DEG; }          // support1: fixed degree, sorted
    for (int j = s; j < e; ++j) {
        float v = cvals[j];
        const float* xr = x + (size_t)cidx[j] * FB;
        if (j + 1 < e)  // gfx1250 global_prefetch_b8 on the next gathered row
            __builtin_prefetch(x + (size_t)cidx[j + 1] * FB + t, 0, 0);
        #pragma unroll
        for (int k = 0; k < 6; ++k) acc[k] += v * xr[t + 352 * k];
    }
    float* yr = y + (size_t)r * FB;
    if (zsub) {
        const float* zr = zsub + (size_t)r * FB;
        #pragma unroll
        for (int k = 0; k < 6; ++k) yr[t + 352 * k] = alpha * acc[k] - zr[t + 352 * k];
    } else {
        #pragma unroll
        for (int k = 0; k < 6; ++k) yr[t + 352 * k] = alpha * acc[k];
    }
}

// ================= WMMA GEMM: (ROWS x 330) @ (330 x OC) + fused epilogue ========
// block = 256 threads = 8 waves; block tile 64 rows x OC cols.
// Full 64x330 fp32 A panel staged into LDS ONCE; single __syncthreads total.
// LDS row layout is UNPADDED in k' (off = k', row stride 362 floats) so every
// fragment ds_load_b64 is base + compile-time immediate; the lane-dependent
// K+8 half-wave shift is absorbed into the base pointer once (hi*8 floats).
// Staging still avoids divisions: the m-loop is unrolled -> m*66 is constant.
// k' in [330,362) zero-filled by the same threads that stage the row (no race).
// B fragments load directly from pre-swizzled global (L2-resident b128 pairs).
// MODE 0: value = sigmoid(acc + b_ru)           (OC = 128)
// MODE 1: out   = u*hx + (1-u)*tanh(acc + b_c)  (OC = 64), u from value[:,64:]
template <int OC, int MODE>
__global__ __launch_bounds__(256) void gemm_kernel(
    const float* __restrict__ xs0, const float* __restrict__ xs1,
    const float* __restrict__ xs2, const float* __restrict__ xs3,
    const float* __restrict__ xs4,
    const __bf16* __restrict__ WbF, const float* __restrict__ bias,
    float* __restrict__ value, const float* __restrict__ hx,
    float* __restrict__ out) {

    constexpr int NT  = OC / 16;   // n-tiles per block (8 or 4)
    constexpr int TPW = NT / 2;    // n-tiles per wave  (4 or 2)
    constexpr int RS  = 362;       // LDS row stride (floats): 8B-aligned rows,
                                   // bank stride 42 -> conflict-free in 16-row strips

    __shared__ __attribute__((aligned(16))) float As[64 * RS];

    int tid  = threadIdx.x;
    int lane = tid & 31;
    int w    = tid >> 5;
    int wm   = w & 3;       // 16-row strip
    int wn   = w >> 2;      // column half
    int rowBase = blockIdx.x * 64;

    // ---- stage full 64 x 330 fp32 A panel + zero pad, m unrolled ----
    {
        int row  = tid >> 2;
        int part = tid & 3;
        int f0   = part * 16;               // parts cover f {0..15,16..31,32..47,48..65}
        int gr   = rowBase + row;
        int b    = gr / NN, n = gr - b * NN;
        size_t gbase = (size_t)n * FB + b * FDIM + f0;
        float* rowDst = &As[row * RS];
        #pragma unroll
        for (int m = 0; m < MMAT; ++m) {
            const float* p = (m == 0) ? xs0 : (m == 1) ? xs1
                           : (m == 2) ? xs2 : (m == 3) ? xs3 : xs4;
            const float* src = p + gbase;
            float* dst = rowDst + m * FDIM + f0;     // m*66 is compile-time
            #pragma unroll
            for (int q = 0; q < 16; ++q) dst[q] = src[q];
            if (part == 3) { dst[16] = src[16]; dst[17] = src[17]; }
        }
        // zero pad k' in [330,362): 8 floats per part, disjoint, race-free
        float* pz = rowDst + FM + part * 8;
        #pragma unroll
        for (int q = 0; q < 8; ++q) pz[q] = 0.f;
    }
    __syncthreads();

    v8f acc[TPW];
    #pragma unroll
    for (int t = 0; t < TPW; ++t) acc[t] = (v8f){0.f,0.f,0.f,0.f,0.f,0.f,0.f,0.f};

    int rowA = wm * 16 + (lane & 15);
    int hi   = lane >> 4;                    // half-wave K+8 shift, folded into base
    const float* aBase = &As[rowA * RS + hi * 8];

    #pragma unroll
    for (int kt = 0; kt < KTILES; ++kt) {
        // ---- A fragment: 8 ds_load_b64 at compile-time immediates + cvt_pk ----
        union { v16bf v; unsigned u[8]; } af;
        #pragma unroll
        for (int j = 0; j < 8; ++j) {
            int kLo = kt * 32 + 2 * (j & 3) + ((j >> 2) << 4);  // compile-time
            union { __bf16 h[2]; unsigned u; } pk;
            pk.h[0] = (__bf16)aBase[kLo];
            pk.h[1] = (__bf16)aBase[kLo + 1];
            af.u[j] = pk.u;
        }
        // ---- B fragments straight from pre-swizzled global (32B/lane each) ----
        #pragma unroll
        for (int t = 0; t < TPW; ++t) {
            int ntg = wn * TPW + t;
            union { v16bf v; uint4 q[2]; } bf;
            const uint4* bp = reinterpret_cast<const uint4*>(
                WbF + (((size_t)kt * NT + ntg) * 32 + lane) * 16);
            bf.q[0] = bp[0];
            bf.q[1] = bp[1];
            acc[t] = __builtin_amdgcn_wmma_f32_16x16x32_bf16(
                false, af.v, false, bf.v, (short)0, acc[t], false, false);
        }
    }

    // ---- fused epilogue (C/D layout: vgpr v -> M = v + 8*(lane>=16), N = lane&15) ----
    int rbase = rowBase + wm * 16 + (hi << 3);
    #pragma unroll
    for (int t = 0; t < TPW; ++t) {
        int gcol = (wn * TPW + t) * 16 + (lane & 15);
        float bia = bias[gcol];
        #pragma unroll
        for (int v = 0; v < 8; ++v) {
            int gr = rbase + v;
            float s = acc[t][v] + bia;
            if (MODE == 0) {
                value[(size_t)gr * (2 * UNITS) + gcol] = 1.0f / (1.0f + __expf(-s));
            } else {
                float cc = tanhf(s);
                float u  = value[(size_t)gr * (2 * UNITS) + UNITS + gcol];
                int b = gr / NN, n = gr - b * NN;
                int oi = b * (NN * UNITS) + n * UNITS + gcol;
                float h = hx[oi];
                out[oi] = u * h + (1.f - u) * cc;
            }
        }
    }
}

// ======================= host orchestration =======================
extern "C" void kernel_launch(void* const* d_in, const int* in_sizes, int n_in,
                              void* d_out, int out_size, void* d_ws, size_t ws_size,
                              hipStream_t stream) {
    const float* inputs = (const float*)d_in[0];
    const float* hx     = (const float*)d_in[1];
    const float* W_ru   = (const float*)d_in[2];
    const float* b_ru   = (const float*)d_in[3];
    const float* W_c    = (const float*)d_in[4];
    const float* b_c    = (const float*)d_in[5];
    const int*   s0r    = (const int*)d_in[6];
    const int*   s0c    = (const int*)d_in[7];
    const float* s0v    = (const float*)d_in[8];
    /* s1_rows (d_in[9]) implicit: rows = i/8 */
    const int*   s1c    = (const int*)d_in[10];
    const float* s1v    = (const float*)d_in[11];
    float* out = (float*)d_out;

    char* ws = (char*)d_ws;
    size_t off = 0;
    auto take = [&](size_t bytes) -> char* {
        char* p = ws + off;
        off += (bytes + 255) & ~(size_t)255;
        return p;
    };
    const size_t XS = (size_t)NN * FB;                 // 16,896,000 floats
    float* x0  = (float*)take(XS * 4);
    float* p1  = (float*)take(XS * 4);
    float* p2  = (float*)take(XS * 4);
    float* p3  = (float*)take(XS * 4);
    float* p4  = (float*)take(XS * 4);
    float* val = (float*)take((size_t)ROWS * 2 * UNITS * 4);   // sigmoid gates (r|u)
    __bf16* WbRU = (__bf16*)take((size_t)KPAD * 2 * UNITS * 2);
    __bf16* WbC  = (__bf16*)take((size_t)KPAD * UNITS * 2);
    int*   cnt   = (int*)take(NN * 4);
    int*   rptr  = (int*)take((NN + 1) * 4);
    int*   ccols = (int*)take(NNZ * 4);
    float* cvals = (float*)take(NNZ * 4);
    (void)ws_size; (void)in_sizes; (void)n_in; (void)out_size;

    // --- CSR for support0 (scattered COO rows -> gather SPMM) ---
    hipMemsetAsync(cnt, 0, NN * 4, stream);
    count_kernel<<<(NNZ + 255) / 256, 256, 0, stream>>>(s0r, cnt);
    scan_kernel<<<1, 1024, 0, stream>>>(cnt, rptr);
    hipMemsetAsync(cnt, 0, NN * 4, stream);
    fill_kernel<<<(NNZ + 255) / 256, 256, 0, stream>>>(s0r, s0c, s0v, rptr, cnt, ccols, cvals);

    // --- permute + pad + swizzle weights into B-fragment layout ---
    convW_kernel<128><<<(KPAD * 2 * UNITS + 255) / 256, 256, 0, stream>>>(W_ru, WbRU);
    convW_kernel<64><<<(KPAD * UNITS + 255) / 256, 256, 0, stream>>>(W_c, WbC);

    const int xGrid = (int)(XS / 256);                 // 66000 exactly

    // ================= gconv 1 (r,u gates) =================
    build_x0_a<<<xGrid, 256, 0, stream>>>(inputs, hx, x0);
    spmm_kernel<<<NN, 352, 0, stream>>>(rptr, ccols, cvals, x0, nullptr, 1.f, p1);
    spmm_kernel<<<NN, 352, 0, stream>>>(rptr, ccols, cvals, p1, x0, 2.f, p2);
    spmm_kernel<<<NN, 352, 0, stream>>>(nullptr, s1c, s1v, x0, nullptr, 1.f, p3);
    spmm_kernel<<<NN, 352, 0, stream>>>(nullptr, s1c, s1v, p3, x0, 2.f, p4);
    gemm_kernel<128, 0><<<ROWS / 64, 256, 0, stream>>>(
        x0, p1, p2, p3, p4, WbRU, b_ru, val, hx, nullptr);

    // ================= gconv 2 (candidate c) + GRU combine =================
    build_x0_b<<<xGrid, 256, 0, stream>>>(inputs, val, hx, x0);
    spmm_kernel<<<NN, 352, 0, stream>>>(rptr, ccols, cvals, x0, nullptr, 1.f, p1);
    spmm_kernel<<<NN, 352, 0, stream>>>(rptr, ccols, cvals, p1, x0, 2.f, p2);
    spmm_kernel<<<NN, 352, 0, stream>>>(nullptr, s1c, s1v, x0, nullptr, 1.f, p3);
    spmm_kernel<<<NN, 352, 0, stream>>>(nullptr, s1c, s1v, p3, x0, 2.f, p4);
    gemm_kernel<64, 1><<<ROWS / 64, 256, 0, stream>>>(
        x0, p1, p2, p3, p4, WbC, b_c, val, hx, out);
}
